// RNNModelSimple_2680059593090
// MI455X (gfx1250) — matrix operand order repaired and checked by reading, not compiled, not run on hardware
//
#include <hip/hip_runtime.h>

#define T_LEN 256
#define B_SZ  32
#define V_SZ  8192
#define H_SZ  1024
#define H4    4096

typedef __attribute__((ext_vector_type(16))) __bf16 v16bf;
typedef __attribute__((ext_vector_type(8)))  float  v8f;

union Frag32B { v16bf v; uint4 q[2]; };

__device__ inline v8f wmma_bf16(v16bf a, v16bf b, v8f c) {
  // D = A(16x32 bf16) x B(32x16 bf16) + C(16x16 f32)
  return __builtin_amdgcn_wmma_f32_16x16x32_bf16(false, a, false, b, (short)0, c,
                                                 false, false);
}

__device__ inline unsigned short f2bf(float f) {
  unsigned int u = __float_as_uint(f);
  u += 0x7fffu + ((u >> 16) & 1u);   // round-to-nearest-even
  return (unsigned short)(u >> 16);
}

__device__ inline float sigmoidf_(float x) { return 1.0f / (1.0f + __expf(-x)); }

// CDNA5 async global->LDS copy (ASYNCcnt-tracked). GV addressing mode:
// VDST = LDS byte offset (low 32 bits of the shared-aperture address),
// VADDR = 64-bit global address.
__device__ inline void async_copy_b128(unsigned lds_addr, const void* gptr) {
  asm volatile("global_load_async_to_lds_b128 %0, %1, off"
               :
               : "v"(lds_addr), "v"((unsigned long long)(uintptr_t)gptr)
               : "memory");
}

__device__ inline unsigned lds_off(const void* p) {
  return (unsigned)(uintptr_t)p;   // LDS aperture: offset lives in addr[31:0]
}

// ---------------------------------------------------------------------------
// Pack a row-major f32 weight [ntiles*16][1024] into bf16 WMMA B-fragment
// layout: out[ntile][kt][lane][16], where lane = half*16 + n,
// element i of the v16bf is K = half*16 + i  (contiguous k order, matching the A fragment loads).
// Each packed fragment is then a single coalesced 32B load per lane.
// ---------------------------------------------------------------------------
__global__ __launch_bounds__(256) void pack_b_kernel(
    const float* __restrict__ W, unsigned short* __restrict__ out, int ntiles) {
  int gid   = blockIdx.x * 256 + threadIdx.x;
  int lane  = gid & 31;
  int kt    = (gid >> 5) & 31;
  int ntile = gid >> 10;
  if (ntile >= ntiles) return;
  int row   = ntile * 16 + (lane & 15);          // output column n
  int kbase = kt * 32 + ((lane >> 4) << 4);      // half*16
  const float* src = W + (size_t)row * H_SZ + kbase;
  alignas(16) unsigned short tmp[16];
#pragma unroll
  for (int i = 0; i < 16; ++i) tmp[i] = f2bf(src[i]);
  uint4* dst = (uint4*)(out + (((size_t)ntile * 32 + kt) * 32 + lane) * 16);
  dst[0] = *(const uint4*)&tmp[0];
  dst[1] = *(const uint4*)&tmp[8];
}

// x_proj[t][b][c] = W_ih[c, X[b,t]] + b_ih[c] + b_hh[c]
__global__ __launch_bounds__(256) void gather_xproj_kernel(
    const int* __restrict__ X, const float* __restrict__ W_ih,
    const float* __restrict__ b_ih, const float* __restrict__ b_hh,
    float* __restrict__ xproj) {
  int gid = blockIdx.x * 256 + threadIdx.x;   // T*B*1024 threads, 4 cols each
  int c0  = (gid & 1023) << 2;
  int b   = (gid >> 10) & 31;
  int t   = gid >> 15;
  int tok = X[b * T_LEN + t];
  const float* base = W_ih + (size_t)c0 * V_SZ + tok;
  float4 r;
  r.x = base[0 * (size_t)V_SZ] + b_ih[c0 + 0] + b_hh[c0 + 0];
  r.y = base[1 * (size_t)V_SZ] + b_ih[c0 + 1] + b_hh[c0 + 1];
  r.z = base[2 * (size_t)V_SZ] + b_ih[c0 + 2] + b_hh[c0 + 2];
  r.w = base[3 * (size_t)V_SZ] + b_ih[c0 + 3] + b_hh[c0 + 3];
  *(float4*)(xproj + ((size_t)t * B_SZ + b) * H4 + c0) = r;
}

__global__ __launch_bounds__(256) void cast_h0_kernel(
    const float* __restrict__ h0, unsigned short* __restrict__ hinit) {
  int i = blockIdx.x * 256 + threadIdx.x;
  if (i < B_SZ * H_SZ) hinit[i] = f2bf(h0[i]);
}

// ---------------------------------------------------------------------------
// One LSTM timestep: gates = xp_t + h_prev @ W_hh^T, then elementwise update.
// Grid: 32 WGs x 256 threads. WG owns 32 hidden units j0..j0+31 (all 4 gates
// for those units -> gate fusion stays inside one WG via LDS).
// Wave w: gate = w>>1 (i,f,g,o), mtile = w&1 (batch rows). Each wave does
// 2 N-tiles x 32 K-steps of v_wmma_f32_16x16x32_bf16.
// ---------------------------------------------------------------------------
__global__ __launch_bounds__(256) void lstm_step_kernel(
    const float* __restrict__ xp_t,              // [B][4H] f32
    const unsigned short* __restrict__ h_prev,   // [B][H] bf16 row-major
    const unsigned short* __restrict__ whh_p,    // packed B fragments
    float* __restrict__ c,                       // [B][H] f32 (in/out)
    float* __restrict__ hf32,                    // [B][H] f32 (out)
    unsigned short* __restrict__ ys_t) {         // [B][H] bf16 (out)
  __shared__ float ldsg[4][32][32];              // gate, batch row, local col
  const int tid   = threadIdx.x;
  const int lane  = tid & 31;
  const int w     = tid >> 5;
  const int gate  = w >> 1;
  const int mtile = w & 1;
  const int half  = lane >> 4;
  const int l15   = lane & 15;
  const int j0    = blockIdx.x * 32;

  v8f acc0 = {0.f, 0.f, 0.f, 0.f, 0.f, 0.f, 0.f, 0.f};
  v8f acc1 = {0.f, 0.f, 0.f, 0.f, 0.f, 0.f, 0.f, 0.f};
  const int ntile0 = gate * 64 + blockIdx.x * 2;

  for (int kt = 0; kt < 32; ++kt) {
    // A fragment (h_prev, row-major bf16): two b128 loads per lane
    Frag32B a;
    const uint4* pa = (const uint4*)(h_prev + (size_t)(mtile * 16 + l15) * H_SZ +
                                     kt * 32 + half * 16);
    a.q[0] = pa[0];
    a.q[1] = pa[1];   // +8 bf16 = +1 uint4
    // B fragments (pre-packed): one contiguous 32B load per lane each
    Frag32B b0, b1;
    const uint4* pb0 =
        (const uint4*)(whh_p + (((size_t)ntile0 * 32 + kt) * 32 + lane) * 16);
    b0.q[0] = pb0[0];
    b0.q[1] = pb0[1];
    const uint4* pb1 = (const uint4*)(whh_p +
        (((size_t)(ntile0 + 1) * 32 + kt) * 32 + lane) * 16);
    b1.q[0] = pb1[0];
    b1.q[1] = pb1[1];
    acc0 = wmma_bf16(a.v, b0.v, acc0);
    acc1 = wmma_bf16(a.v, b1.v, acc1);
  }

  // Epilogue: add x_proj, activate, stage into LDS for the gate fusion
#pragma unroll
  for (int r = 0; r < 8; ++r) {
    int m = mtile * 16 + half * 8 + r;
    {
      int col   = gate * H_SZ + j0 + l15;
      float v   = acc0[r] + xp_t[(size_t)m * H4 + col];
      ldsg[gate][m][l15] = (gate == 2) ? tanhf(v) : sigmoidf_(v);
    }
    {
      int col   = gate * H_SZ + j0 + 16 + l15;
      float v   = acc1[r] + xp_t[(size_t)m * H4 + col];
      ldsg[gate][m][16 + l15] = (gate == 2) ? tanhf(v) : sigmoidf_(v);
    }
  }
  __syncthreads();

  // Elementwise c/h update: 32 rows x 32 local cols = 1024 elems, 4 per thread
#pragma unroll
  for (int k = 0; k < 4; ++k) {
    int e  = k * 256 + tid;
    int m  = e >> 5;
    int jl = e & 31;
    int j  = j0 + jl;
    float iv = ldsg[0][m][jl];
    float fv = ldsg[1][m][jl];
    float gv = ldsg[2][m][jl];
    float ov = ldsg[3][m][jl];
    size_t idx = (size_t)m * H_SZ + j;
    float cn = fv * c[idx] + iv * gv;
    float hn = ov * tanhf(cn);
    c[idx]    = cn;
    hf32[idx] = hn;
    ys_t[idx] = f2bf(hn);
  }
}

// ---------------------------------------------------------------------------
// Output GEMM: Y[8192x8192] = Ys[8192x1024](bf16) @ W_out^T(bf16 packed)+b_out
// 128x128 block per WG, 8 waves, each wave 64x32 = 8 WMMA tiles.
// K=64 chunks double-buffered in LDS via CDNA5 async global->LDS copies
// (ASYNCcnt): 16 WMMAs per barrier pair, no intra-WG redundant L2 fetches.
// LDS: 2 x (16KB A + 16KB B) = 64KB.
// ---------------------------------------------------------------------------
__global__ __launch_bounds__(256) void gemm_out_kernel(
    const unsigned short* __restrict__ Ys,     // [8192][1024] bf16
    const unsigned short* __restrict__ Wp,     // packed [512][32][32][16]
    const float* __restrict__ b_out, float* __restrict__ Y) {
  __shared__ unsigned short Abuf[2][128][64];     // rows x K64 chunk
  __shared__ unsigned short Bbuf[2][8][2][512];   // ntile x kt-sub x frag
  const int tid  = threadIdx.x;
  const int lane = tid & 31;
  const int w    = tid >> 5;
  const int half = lane >> 4, l15 = lane & 15;
  const int bm = blockIdx.x * 128;
  const int bn = blockIdx.y * 128;
  const int wm = (w & 1) * 64;
  const int wn = (w >> 1) * 32;
  const int ntb0 = bn >> 4;                    // first of 8 ntiles in block

  // per-thread copy assignment: 64B of A + 64B of B per K64 stage
  const int arow = tid >> 1;
  const int acol = (tid & 1) * 32;
  const int bntl = tid >> 5;
  const int binr = tid & 31;

  v8f acc[4][2];
#pragma unroll
  for (int mt = 0; mt < 4; ++mt)
#pragma unroll
    for (int ntl = 0; ntl < 2; ++ntl)
      acc[mt][ntl] = (v8f){0.f, 0.f, 0.f, 0.f, 0.f, 0.f, 0.f, 0.f};

  // stage K64 chunk at K offset `ko` into LDS buffer `buf`
  // (8 async b128 per thread)
  auto stage = [&](int buf, int ko) {
    const unsigned short* ga = Ys + (size_t)(bm + arow) * H_SZ + ko + acol;
#pragma unroll
    for (int i = 0; i < 4; ++i)
      async_copy_b128(lds_off(&Abuf[buf][arow][acol + 8 * i]), ga + 8 * i);
#pragma unroll
    for (int kk = 0; kk < 2; ++kk) {
      const unsigned short* gb =
          Wp + (((size_t)(ntb0 + bntl) * 32 + (ko >> 5) + kk) * 512) +
          binr * 16;
      async_copy_b128(lds_off(&Bbuf[buf][bntl][kk][binr * 16]), gb);
      async_copy_b128(lds_off(&Bbuf[buf][bntl][kk][binr * 16 + 8]), gb + 8);
    }
  };

  stage(0, 0);
  for (int ko = 0; ko < H_SZ; ko += 64) {
    const int cur = (ko >> 6) & 1;
    if (ko + 64 < H_SZ) {
      stage(cur ^ 1, ko + 64);
      // async loads retire in order: allowing 8 outstanding (the 8 just
      // issued for the next buffer) guarantees the current buffer is done.
      asm volatile("s_wait_asynccnt 0x8" ::: "memory");
    } else {
      asm volatile("s_wait_asynccnt 0x0" ::: "memory");
    }
    __syncthreads();

#pragma unroll
    for (int kk = 0; kk < 2; ++kk) {
      Frag32B a[4];
#pragma unroll
      for (int mt = 0; mt < 4; ++mt) {
        const uint4* pa =
            (const uint4*)(&Abuf[cur][wm + mt * 16 + l15][kk * 32 + half * 16]);
        a[mt].q[0] = pa[0];
        a[mt].q[1] = pa[1];
      }
      Frag32B bf[2];
#pragma unroll
      for (int ntl = 0; ntl < 2; ++ntl) {
        const uint4* pb =
            (const uint4*)(&Bbuf[cur][(w >> 1) * 2 + ntl][kk][lane * 16]);
        bf[ntl].q[0] = pb[0];
        bf[ntl].q[1] = pb[1];
      }
#pragma unroll
      for (int mt = 0; mt < 4; ++mt)
#pragma unroll
        for (int ntl = 0; ntl < 2; ++ntl)
          acc[mt][ntl] = wmma_bf16(a[mt].v, bf[ntl].v, acc[mt][ntl]);
    }

    __syncthreads();
  }

#pragma unroll
  for (int ntl = 0; ntl < 2; ++ntl) {
    int col  = bn + wn + ntl * 16 + l15;
    float bo = b_out[col];
#pragma unroll
    for (int mt = 0; mt < 4; ++mt) {
#pragma unroll
      for (int r = 0; r < 8; ++r) {
        int row = bm + wm + mt * 16 + half * 8 + r;
        Y[(size_t)row * V_SZ + col] = acc[mt][ntl][r] + bo;
      }
    }
  }
}

__global__ __launch_bounds__(256) void finalize_kernel(
    const float* __restrict__ hf32, const float* __restrict__ c,
    float* __restrict__ out_tail) {
  int i = blockIdx.x * 256 + threadIdx.x;
  if (i < B_SZ * H_SZ) {
    out_tail[i] = hf32[i];                  // hT
    out_tail[B_SZ * H_SZ + i] = c[i];       // cT
  }
}

extern "C" void kernel_launch(void* const* d_in, const int* in_sizes, int n_in,
                              void* d_out, int out_size, void* d_ws,
                              size_t ws_size, hipStream_t stream) {
  (void)in_sizes; (void)n_in; (void)out_size; (void)ws_size;
  const int*   X     = (const int*)d_in[0];
  const float* h0    = (const float*)d_in[1];
  const float* c0    = (const float*)d_in[2];
  const float* W_ih  = (const float*)d_in[3];
  const float* W_hh  = (const float*)d_in[4];
  const float* b_ih  = (const float*)d_in[5];
  const float* b_hh  = (const float*)d_in[6];
  const float* W_out = (const float*)d_in[7];
  const float* b_out = (const float*)d_in[8];
  float* Y = (float*)d_out;

  char* ws = (char*)d_ws;
  size_t off = 0;
  float* xproj = (float*)(ws + off);            off += (size_t)T_LEN * B_SZ * H4 * 4;
  unsigned short* whh_p  = (unsigned short*)(ws + off); off += (size_t)H4 * H_SZ * 2;
  unsigned short* wout_p = (unsigned short*)(ws + off); off += (size_t)V_SZ * H_SZ * 2;
  unsigned short* Ys     = (unsigned short*)(ws + off); off += (size_t)T_LEN * B_SZ * H_SZ * 2;
  unsigned short* hinit  = (unsigned short*)(ws + off); off += (size_t)B_SZ * H_SZ * 2;
  float* cbuf = (float*)(ws + off);             off += (size_t)B_SZ * H_SZ * 4;
  float* hf32 = (float*)(ws + off);             off += (size_t)B_SZ * H_SZ * 4;

  // One-time packs / casts (memory-bound, L2-resident weights)
  pack_b_kernel<<<(256 * 32 * 32) / 256, 256, 0, stream>>>(W_hh, whh_p, 256);
  pack_b_kernel<<<(512 * 32 * 32) / 256, 256, 0, stream>>>(W_out, wout_p, 512);
  cast_h0_kernel<<<(B_SZ * H_SZ + 255) / 256, 256, 0, stream>>>(h0, hinit);
  hipMemcpyAsync(cbuf, c0, (size_t)B_SZ * H_SZ * 4, hipMemcpyDeviceToDevice,
                 stream);
  gather_xproj_kernel<<<(T_LEN * B_SZ * 1024) / 256, 256, 0, stream>>>(
      X, W_ih, b_ih, b_hh, xproj);

  // Serial recurrence: 256 graph-captured step launches
  for (int t = 0; t < T_LEN; ++t) {
    const unsigned short* hp =
        (t == 0) ? hinit : (Ys + (size_t)(t - 1) * B_SZ * H_SZ);
    lstm_step_kernel<<<32, 256, 0, stream>>>(
        xproj + (size_t)t * B_SZ * H4, hp, whh_p, cbuf, hf32,
        Ys + (size_t)t * B_SZ * H_SZ);
  }

  // Output projection
  gemm_out_kernel<<<dim3(64, 64), 256, 0, stream>>>(Ys, wout_p, b_out, Y);
  finalize_kernel<<<(B_SZ * H_SZ + 255) / 256, 256, 0, stream>>>(
      hf32, cbuf, Y + (size_t)T_LEN * B_SZ * V_SZ);
}
